// HeteroVGAE_38225208934585
// MI455X (gfx1250) — matrix operand order repaired and checked by reading, not compiled, not run on hardware
//
#include <hip/hip_runtime.h>
#include <hip/hip_bf16.h>
#include <stdint.h>

// ---------------------------------------------------------------------------
// HeteroVGAE forward for MI455X (gfx1250, wave32, WMMA).
// Sizes: N_DIS = N_GENE = 20000, E = 640000, IN_C = 128, OUT_C = 64.
//
// Pipeline:
//   1. zero agg/cnt
//   2. wave-per-edge scatter-add (L2-resident f32 atomics)
//   3. streaming f32->f16 conversion of x_dis, x_gene, and agg/cnt -> mean
//   4. fused WMMA node kernel (SAGE layer + mu/logvar heads + reparameterize)
//      - A fragments: f16 global_load_b128, no conversion VALU in hot loop
//      - B fragments: f16 weights staged once per block in LDS, transposed
//   5. wave-per-edge inner-product decode + sigmoid
// ---------------------------------------------------------------------------

#define N_NODES    20000
#define IN_C       128
#define OUT_C      64
#define TILE_M     16          // one WMMA row tile
#define TILES_PB   5           // row tiles per block -> 80 rows/block
#define ROWS_PB    (TILE_M * TILES_PB)
#define WAVES_PB   4           // 4 waves -> 4 x 16 = 64 output columns
#define LDK_W      (IN_C + 8)  // padded LDS row stride (halves) for W_l/W_r
#define LDK_H      (OUT_C + 8) // padded LDS row stride (halves) for heads

typedef __attribute__((ext_vector_type(16))) _Float16 v16h;
typedef __attribute__((ext_vector_type(8)))  _Float16 v8h;
typedef __attribute__((ext_vector_type(4)))  _Float16 v4h;
typedef __attribute__((ext_vector_type(8)))  float    v8f;

__device__ inline v8f wmma_f16(v16h a, v16h b, v8f c) {
  return __builtin_amdgcn_wmma_f32_16x16x32_f16(
      /*neg_a=*/false, a, /*neg_b=*/false, b,
      /*c_mod=*/(short)0, c, /*reuse_a=*/false, /*reuse_b=*/false);
}

// ---------------- WMMA fragment loaders (CDNA5 §7.12.2 layouts) -------------

// A: 16x32 f16 from a row-major f16 source (global or LDS), stride ld halves.
// Lane (m = lane&15) supplies row `row`; g = lane>>4.
// v16h elements 0..7  <- K = k0 + 8g   .. k0 + 8g+7    (8 contiguous halves)
// v16h elements 8..15 <- K = k0+16+8g .. k0+16+8g+7    (8 contiguous halves)
__device__ inline v16h load_a_f16(const _Float16* src, int row, int k0,
                                  int ld, int g) {
  const _Float16* p = src + (size_t)row * ld + k0 + 8 * g;
  const v8h lo = *(const v8h*)(p);        // 16B load
  const v8h hi = *(const v8h*)(p + 16);   // 16B load
  v16h a;
#pragma unroll
  for (int j = 0; j < 8; ++j) { a[j] = lo[j]; a[8 + j] = hi[j]; }
  return a;
}

// B: 32x16 f16 from LDS weights stored transposed as wt[n][k] (stride ldk).
// Lane (n = lane&15) supplies column n0+n; g = lane>>4.
// v16h elements 0..7 <- K = k0 + 8g .. k0 + 8g+7; 8..15 <- K = k0+16+8g .. k0+16+8g+7
__device__ inline v16h load_b_lds(const _Float16* wt, int k0, int n0,
                                  int ldk, int lane) {
  const int n = lane & 15, g = lane >> 4;
  const _Float16* p = wt + (size_t)(n0 + n) * ldk + k0 + 8 * g;
  const v8h lo = *(const v8h*)(p);        // ds_load_b128
  const v8h hi = *(const v8h*)(p + 16);   // ds_load_b128
  v16h b;
#pragma unroll
  for (int j = 0; j < 8; ++j) { b[j] = lo[j]; b[8 + j] = hi[j]; }
  return b;
}

// ---------------- hash-based standard normal (Box-Muller) -------------------
__device__ inline float hash_normal(uint32_t x, uint32_t y) {
  uint32_t s = x * 0x9E3779B9u ^ (y + 0x7F4A7C15u) * 0x85EBCA6Bu;
  s ^= s >> 16; s *= 0x7FEB352Du; s ^= s >> 15; s *= 0x846CA68Bu; s ^= s >> 16;
  uint32_t t = s * 0xC2B2AE35u + 0x165667B1u;
  t ^= t >> 13; t *= 0x5BD1E995u; t ^= t >> 15;
  float u1 = (float)(s >> 8) * (1.0f / 16777216.0f) + 1.0e-7f;
  float u2 = (float)(t >> 8) * (1.0f / 16777216.0f);
  return sqrtf(-2.0f * logf(u1)) * cosf(6.28318530718f * u2);
}

// ---------------- kernel 1: zero agg + cnt ---------------------------------
__global__ void k_zero(float* __restrict__ p, int n) {
  int i = blockIdx.x * blockDim.x + threadIdx.x;
  int stride = gridDim.x * blockDim.x;
  for (; i < n; i += stride) p[i] = 0.0f;
}

// ---------------- kernel 2: scatter mean numerator/denominator -------------
// One wave32 per edge; each lane atomically adds 4 contiguous channels.
// x_disease (10 MB) and agg (10 MB) are L2-resident (192 MB L2).
__global__ void k_scatter(const float* __restrict__ xd,
                          const int* __restrict__ src,
                          const int* __restrict__ dst,
                          float* __restrict__ agg,
                          float* __restrict__ cnt, int nedges) {
  const int lane = threadIdx.x & 31;
  int e = (blockIdx.x * blockDim.x + threadIdx.x) >> 5;
  const int nwaves = (gridDim.x * blockDim.x) >> 5;
  for (; e < nedges; e += nwaves) {
    const int s = src[e], d = dst[e];
    const float4 v = ((const float4*)(xd + (size_t)s * IN_C))[lane];
    float* ap = agg + (size_t)d * IN_C + lane * 4;
    atomicAdd(ap + 0, v.x);
    atomicAdd(ap + 1, v.y);
    atomicAdd(ap + 2, v.z);
    atomicAdd(ap + 3, v.w);
    if (lane == 0) atomicAdd(cnt + d, 1.0f);
  }
}

// ---------------- kernel 3a: streaming f32 -> f16 conversion ---------------
// One thread per 4 elements: float4 load -> half4 store (fully coalesced).
__global__ void k_cvt(const float* __restrict__ src, _Float16* __restrict__ dst,
                      int nquads) {
  int i = blockIdx.x * blockDim.x + threadIdx.x;
  int stride = gridDim.x * blockDim.x;
  for (; i < nquads; i += stride) {
    const float4 v = ((const float4*)src)[i];
    v4h h; h[0] = (_Float16)v.x; h[1] = (_Float16)v.y;
           h[2] = (_Float16)v.z; h[3] = (_Float16)v.w;
    ((v4h*)dst)[i] = h;
  }
}

// ---------------- kernel 3b: agg/cnt -> normalized f16 mean ----------------
__global__ void k_mean(const float* __restrict__ agg,
                       const float* __restrict__ cnt,
                       _Float16* __restrict__ dst, int nquads) {
  int i = blockIdx.x * blockDim.x + threadIdx.x;
  int stride = gridDim.x * blockDim.x;
  for (; i < nquads; i += stride) {
    const int row = (i * 4) >> 7;                 // IN_C = 128 -> 32 quads/row
    const float rc = 1.0f / fmaxf(cnt[row], 1.0f);
    const float4 v = ((const float4*)agg)[i];
    v4h h; h[0] = (_Float16)(v.x * rc); h[1] = (_Float16)(v.y * rc);
           h[2] = (_Float16)(v.z * rc); h[3] = (_Float16)(v.w * rc);
    ((v4h*)dst)[i] = h;
  }
}

// ---------------- kernel 4: fused SAGE layer + VGAE heads (WMMA) -----------
// Block = 128 threads = 4 waves; wave w owns output columns [16w, 16w+16).
// Block b owns rows [80b, 80b+80) processed as five 16-row WMMA tiles.
//   gene:    h = mean @ W_l + x @ W_r + b_l     (mh = pre-normalized f16 mean)
//   disease: h =              x @ W_r + b_l
//   z = (h @ Wmu + bmu) + eps * exp(h @ Wlv + blv)
__global__ __launch_bounds__(TILE_M * WAVES_PB * 2)
void k_node(const _Float16* __restrict__ xh,
            const _Float16* __restrict__ mh,     // null for disease
            const float* __restrict__ W_l,
            const float* __restrict__ b_l,
            const float* __restrict__ W_r,
            const float* __restrict__ Wmu, const float* __restrict__ bmu,
            const float* __restrict__ Wlv, const float* __restrict__ blv,
            float* __restrict__ z,
            int is_gene, uint32_t seed) {
  __shared__ _Float16 wl_t[OUT_C * LDK_W];    // W_l^T  [64][136] f16, 17 KB
  __shared__ _Float16 wr_t[OUT_C * LDK_W];    // W_r^T                 17 KB
  __shared__ _Float16 wm_t[OUT_C * LDK_H];    // Wmu^T  [64][72] f16,   9 KB
  __shared__ _Float16 wv_t[OUT_C * LDK_H];    // Wlv^T                  9 KB
  __shared__ _Float16 h_lds[TILE_M * OUT_C];  // stage-2 A staging (f16), 2 KB

  const int tid  = threadIdx.x;
  const int lane = threadIdx.x & 31;
  const int wid  = threadIdx.x >> 5;          // N-tile index, 0..3
  const int n0   = wid * 16;
  const int ncol = lane & 15;
  const int grow = lane >> 4;                 // 0/1 -> C rows i / i+8
  const int arow = lane & 15;                 // A-fragment row within tile

  // ---- stage weights into LDS (coalesced global reads, f16 transpose) ----
#pragma unroll 4
  for (int i = tid; i < IN_C * OUT_C; i += 128) {
    const int k = i >> 6, n = i & 63;         // W is [K=128][N=64] row-major
    wl_t[n * LDK_W + k] = (_Float16)W_l[i];
    wr_t[n * LDK_W + k] = (_Float16)W_r[i];
  }
#pragma unroll 4
  for (int i = tid; i < OUT_C * OUT_C; i += 128) {
    const int k = i >> 6, n = i & 63;         // heads are [K=64][N=64]
    wm_t[n * LDK_H + k] = (_Float16)Wmu[i];
    wv_t[n * LDK_H + k] = (_Float16)Wlv[i];
  }
  const float bias_l = b_l[n0 + ncol];
  const float bm     = bmu[n0 + ncol];
  const float bv     = blv[n0 + ncol];
  __syncthreads();

  for (int t = 0; t < TILES_PB; ++t) {
    const int row0 = blockIdx.x * ROWS_PB + t * TILE_M;

    // ---- stage 1: h[16,64] ----
    v8f c;
#pragma unroll
    for (int i = 0; i < 8; ++i) c[i] = bias_l;

    if (is_gene) {
#pragma unroll
      for (int k0 = 0; k0 < IN_C; k0 += 32) {
        v16h a = load_a_f16(mh, row0 + arow, k0, IN_C, grow);
        v16h b = load_b_lds(wl_t, k0, n0, LDK_W, lane);
        c = wmma_f16(a, b, c);
      }
    }
#pragma unroll
    for (int k0 = 0; k0 < IN_C; k0 += 32) {
      v16h a = load_a_f16(xh, row0 + arow, k0, IN_C, grow);
      v16h b = load_b_lds(wr_t, k0, n0, LDK_W, lane);
      c = wmma_f16(a, b, c);
    }

    // h tile -> LDS (f16) so every wave sees the full K=64 rows for stage 2
#pragma unroll
    for (int i = 0; i < 8; ++i)
      h_lds[(i + 8 * grow) * OUT_C + n0 + ncol] = (_Float16)c[i];
    __syncthreads();

    // ---- stage 2: mu / logvar heads, K = 64 ----
    v8f cmu, clv;
#pragma unroll
    for (int i = 0; i < 8; ++i) { cmu[i] = bm; clv[i] = bv; }

#pragma unroll
    for (int k0 = 0; k0 < OUT_C; k0 += 32) {
      v16h a  = load_a_f16(h_lds, arow, k0, OUT_C, grow);
      v16h b1 = load_b_lds(wm_t, k0, n0, LDK_H, lane);
      cmu = wmma_f16(a, b1, cmu);
      v16h b2 = load_b_lds(wv_t, k0, n0, LDK_H, lane);
      clv = wmma_f16(a, b2, clv);
    }

    // ---- reparameterize + store z ----
#pragma unroll
    for (int i = 0; i < 8; ++i) {
      const int row = row0 + i + 8 * grow;
      const int col = n0 + ncol;
      const float eps = hash_normal(seed + (uint32_t)row, (uint32_t)col);
      z[(size_t)row * OUT_C + col] = cmu[i] + eps * expf(clv[i]);
    }
    __syncthreads();   // h_lds reused next tile
  }
}

// ---------------- kernel 5: inner-product decode + sigmoid -----------------
// One wave32 per edge: each lane holds float2 of the 64-dim z vectors.
// z_d / z_g (5 MB each) are L2-resident.
__global__ void k_decode(const float* __restrict__ zd,
                         const float* __restrict__ zg,
                         const int* __restrict__ src,
                         const int* __restrict__ dst,
                         float* __restrict__ out, int nedges) {
  const int lane = threadIdx.x & 31;
  int e = (blockIdx.x * blockDim.x + threadIdx.x) >> 5;
  const int nwaves = (gridDim.x * blockDim.x) >> 5;
  for (; e < nedges; e += nwaves) {
    const int s = src[e], d = dst[e];
    const float2 a = ((const float2*)(zd + (size_t)s * OUT_C))[lane];
    const float2 b = ((const float2*)(zg + (size_t)d * OUT_C))[lane];
    float p = a.x * b.x + a.y * b.y;
#pragma unroll
    for (int off = 16; off > 0; off >>= 1) p += __shfl_xor(p, off, 32);
    if (lane == 0) out[e] = 1.0f / (1.0f + expf(-p));
  }
}

// ---------------------------------------------------------------------------
extern "C" void kernel_launch(void* const* d_in, const int* in_sizes, int n_in,
                              void* d_out, int out_size, void* d_ws, size_t ws_size,
                              hipStream_t stream) {
  const float* x_dis   = (const float*)d_in[0];
  const float* x_gene  = (const float*)d_in[1];
  const int*   e_src   = (const int*)d_in[2];
  const int*   e_dst   = (const int*)d_in[3];
  const float* W_l     = (const float*)d_in[4];
  const float* b_l     = (const float*)d_in[5];
  const float* W_r     = (const float*)d_in[6];
  const float* Wmu_d   = (const float*)d_in[7];
  const float* bmu_d   = (const float*)d_in[8];
  const float* Wlv_d   = (const float*)d_in[9];
  const float* blv_d   = (const float*)d_in[10];
  const float* Wmu_g   = (const float*)d_in[11];
  const float* bmu_g   = (const float*)d_in[12];
  const float* Wlv_g   = (const float*)d_in[13];
  const float* blv_g   = (const float*)d_in[14];
  float*       logits  = (float*)d_out;
  const int    nedges  = in_sizes[2];

  // workspace layout:
  //   f32: agg [N*IN_C] | cnt [N] | z_d [N*OUT_C] | z_g [N*OUT_C]
  //   f16: xh_d [N*IN_C] | xh_g [N*IN_C] | mh [N*IN_C]
  float* ws   = (float*)d_ws;
  float* agg  = ws;
  float* cnt  = agg + (size_t)N_NODES * IN_C;
  float* z_d  = cnt + N_NODES;
  float* z_g  = z_d + (size_t)N_NODES * OUT_C;
  _Float16* xh_d = (_Float16*)(z_g + (size_t)N_NODES * OUT_C);
  _Float16* xh_g = xh_d + (size_t)N_NODES * IN_C;
  _Float16* mh   = xh_g + (size_t)N_NODES * IN_C;

  const int nquads = N_NODES * IN_C / 4;      // 640000

  // 1. zero the aggregation buffers (ws is poisoned by the harness)
  {
    const int n = N_NODES * IN_C + N_NODES;
    k_zero<<<(n + 255) / 256, 256, 0, stream>>>(agg, n);
  }

  // 2. scatter-add messages (L2-resident atomics)
  k_scatter<<<4096, 256, 0, stream>>>(x_dis, e_src, e_dst, agg, cnt, nedges);

  // 3. streaming f16 conversions (mean conversion folds the 1/cnt division)
  k_cvt <<<2500, 256, 0, stream>>>(x_dis,  xh_d, nquads);
  k_cvt <<<2500, 256, 0, stream>>>(x_gene, xh_g, nquads);
  k_mean<<<2500, 256, 0, stream>>>(agg, cnt, mh, nquads);

  // 4. fused SAGE + heads + reparameterize (WMMA), per node type
  const int nblocks = N_NODES / ROWS_PB;   // 250
  k_node<<<nblocks, TILE_M * WAVES_PB * 2, 0, stream>>>(
      xh_d, nullptr, W_l, b_l, W_r,
      Wmu_d, bmu_d, Wlv_d, blv_d, z_d, /*is_gene=*/0, 0x0D15EA5Eu);
  k_node<<<nblocks, TILE_M * WAVES_PB * 2, 0, stream>>>(
      xh_g, mh, W_l, b_l, W_r,
      Wmu_g, bmu_g, Wlv_g, blv_g, z_g, /*is_gene=*/1, 0xBADC0FFEu);

  // 5. decode edges
  k_decode<<<4096, 256, 0, stream>>>(z_d, z_g, e_src, e_dst, logits, nedges);
}
